// TriangleMultiplication_51101520887967
// MI455X (gfx1250) — matrix operand order repaired and checked by reading, not compiled, not run on hardware
//
#include <hip/hip_runtime.h>

#define NDIM   512
#define DDIM   128
#define NN     (NDIM * NDIM)      // 262144 rows of the flattened pair tensor
#define LN_EPS 1e-5f

typedef __attribute__((ext_vector_type(16))) __bf16 v16bf;
typedef __attribute__((ext_vector_type(8)))  __bf16 v8bf;
typedef __attribute__((ext_vector_type(8)))  float  v8f;
typedef __attribute__((ext_vector_type(4)))  float  v4f;

union Frag16 { v16bf v; v8bf h[2]; };
union Pack16 { v8bf v[2]; __bf16 s[16]; };
union Pack8  { v8bf v;    __bf16 s[8];  };

// ---------------- CDNA5 async (GLOBAL_LOAD_ASYNC_TO_LDS) plumbing -----------
#if defined(__gfx1250__) && __has_builtin(__builtin_amdgcn_global_load_async_to_lds_b128)
#define USE_ASYNC 1
#else
#define USE_ASYNC 0
#endif

// Builtin prototype (from hipcc diagnostic) uses generic vector-of-4-int
// pointees: arg0 = int4 addrspace(1)* (global src), arg1 = int4 addrspace(3)*
// (LDS dst), then immediate offset and cpol.
typedef int b128_t __attribute__((vector_size(16)));

// Copy 16 bytes (8 bf16) global -> LDS.
static __device__ __forceinline__ void async_copy16(const __bf16* g, __bf16* l) {
#if USE_ASYNC
  __builtin_amdgcn_global_load_async_to_lds_b128(
      (__attribute__((address_space(1))) b128_t*)g,
      (__attribute__((address_space(3))) b128_t*)l, 0, 0);
#else
  *(v8bf*)l = *(const v8bf*)g;
#endif
}

template <int N>
static __device__ __forceinline__ void wait_async() {
#if USE_ASYNC
#if __has_builtin(__builtin_amdgcn_s_wait_asynccnt)
  __builtin_amdgcn_s_wait_asynccnt(N);
#else
  asm volatile("s_wait_asynccnt %0" :: "i"(N) : "memory");
#endif
#endif
}

// ---------------------------------------------------------------------------
static __device__ __forceinline__ v8f wmma_bf16(v16bf a, v16bf b, v8f c) {
  return __builtin_amdgcn_wmma_f32_16x16x32_bf16(false, a, false, b,
                                                 (short)0, c, false, false);
}

// A fragment: 16(M) x 32(K), row-major LDS tile, stride lda (elements).
// lanes 0-15: M=lane, K in {0..7, 16..23}; lanes 16-31: M=lane-16, K in {8..15, 24..31}
static __device__ __forceinline__ v16bf load_a(const __bf16* As, int lda,
                                               int mBase, int kBase, int lane) {
  int lm = lane & 15, hi = lane >> 4;
  const __bf16* p = As + (mBase + lm) * lda + kBase + hi * 8;
  Frag16 f;
  f.h[0] = *(const v8bf*)(p);
  f.h[1] = *(const v8bf*)(p + 16);
  return f.v;
}

// B fragment: 32(K) x 16(N), row-major LDS tile, stride ldb.
// lanes 0-15: N=lane, K in {0..7, 16..23}; lanes 16-31: N=lane-16, K in {8..15, 24..31}
static __device__ __forceinline__ v16bf load_b(const __bf16* Bs, int ldb,
                                               int kBase, int nBase, int lane) {
  int ln = lane & 15, hi = lane >> 4;
  const __bf16* p = Bs + (kBase + hi * 8) * ldb + nBase + ln;
  v16bf v;
  #pragma unroll
  for (int s = 0; s < 8; ++s) {
    v[s] = p[s * ldb];
    v[8 + s] = p[(16 + s) * ldb];
  }
  return v;
}

// ---------------------------------------------------------------------------
// Kernel 1: projection GEMM  out = f(x @ W + bias), scattered into per-channel
// plane layouts (bf16).  mode 0: A plane [c][i][j]; mode 1: B plane [c][j][k];
// mode 2: sigmoid -> G plane [c][i][k].
// Block: 128 rows x 64 cols, 8 waves as 4x2, each wave 32x32 (2x2 WMMA frags).
// (keeps the VGPR path: must convert f32 -> bf16 on the way to LDS)
// ---------------------------------------------------------------------------
__global__ __launch_bounds__(256)
void proj_kernel(const float* __restrict__ x,     // (NN, 128)
                 const float* __restrict__ W,     // (128, 128) row-major (in,out)
                 const float* __restrict__ bias,  // (128)
                 __bf16* __restrict__ out,        // plane buffer (128*NN bf16)
                 int mode) {
  __shared__ __align__(16) __bf16 As[128 * 40];
  __shared__ __align__(16) __bf16 Bs[32 * 72];
  const int tid  = threadIdx.x;
  const int lane = tid & 31, wave = tid >> 5;
  const int wm = wave >> 1, wn = wave & 1;      // 4 x 2 wave grid
  const int m0 = blockIdx.x * 128;
  const int c0 = blockIdx.y * 64;

  v8f acc[2][2] = {};

  for (int k0 = 0; k0 < DDIM; k0 += 32) {
    { // X tile 128x32 f32 -> bf16 LDS
      int r = tid >> 1, half = tid & 1;
      const float* src = x + (size_t)(m0 + r) * DDIM + k0 + half * 16;
      Pack16 pk;
      #pragma unroll
      for (int i = 0; i < 16; i += 4) {
        v4f t = *(const v4f*)(src + i);
        pk.s[i] = (__bf16)t.x; pk.s[i+1] = (__bf16)t.y;
        pk.s[i+2] = (__bf16)t.z; pk.s[i+3] = (__bf16)t.w;
      }
      v8bf* dst = (v8bf*)(As + r * 40 + half * 16);
      dst[0] = pk.v[0]; dst[1] = pk.v[1];
    }
    { // W tile 32x64 f32 -> bf16 LDS
      int kr = tid >> 3, cg = tid & 7;
      const float* src = W + (size_t)(k0 + kr) * DDIM + c0 + cg * 8;
      Pack8 pk;
      #pragma unroll
      for (int i = 0; i < 8; i += 4) {
        v4f t = *(const v4f*)(src + i);
        pk.s[i] = (__bf16)t.x; pk.s[i+1] = (__bf16)t.y;
        pk.s[i+2] = (__bf16)t.z; pk.s[i+3] = (__bf16)t.w;
      }
      *(v8bf*)(Bs + kr * 72 + cg * 8) = pk.v;
    }
    __syncthreads();
    v16bf a0 = load_a(As, 40, wm * 32,      0, lane);
    v16bf a1 = load_a(As, 40, wm * 32 + 16, 0, lane);
    v16bf b0 = load_b(Bs, 72, 0, wn * 32,      lane);
    v16bf b1 = load_b(Bs, 72, 0, wn * 32 + 16, lane);
    acc[0][0] = wmma_bf16(a0, b0, acc[0][0]);
    acc[0][1] = wmma_bf16(a0, b1, acc[0][1]);
    acc[1][0] = wmma_bf16(a1, b0, acc[1][0]);
    acc[1][1] = wmma_bf16(a1, b1, acc[1][1]);
    __syncthreads();
  }

  const int lm = lane & 15, hi = lane >> 4;
  #pragma unroll
  for (int mi = 0; mi < 2; ++mi)
    #pragma unroll
    for (int ni = 0; ni < 2; ++ni) {
      int col = c0 + wn * 32 + ni * 16 + lm;   // output channel c
      float bv = bias[col];
      #pragma unroll
      for (int e = 0; e < 8; ++e) {
        int m = m0 + wm * 32 + mi * 16 + e + 8 * hi;
        float v = acc[mi][ni][e] + bv;
        size_t idx;
        if (mode == 0) {
          // a[j,i,c] -> A[c][i][j]; j = m>>9, i = m&511
          idx = (size_t)col * NN + (size_t)(m & (NDIM - 1)) * NDIM + (m >> 9);
        } else {
          if (mode == 2) v = 1.0f / (1.0f + __expf(-v));
          // b[j,k,c] -> B[c][j][k] and g[i,k,c] -> G[c][i][k]: both = c*NN + m
          idx = (size_t)col * NN + m;
        }
        out[idx] = (__bf16)v;
      }
    }
}

// ---------------------------------------------------------------------------
// Kernel 2: triangle einsum — 128 independent 512x512x512 GEMMs (one per
// channel plane), gated by G in the epilogue, written back in (row, c) layout.
// Block: 128(i) x 64(k) tile, contraction over j in steps of 32.
// Double-buffered LDS fed by GLOBAL_LOAD_ASYNC_TO_LDS_B128 (ASYNCcnt).
// ---------------------------------------------------------------------------
__global__ __launch_bounds__(256)
void tri_kernel(const __bf16* __restrict__ Ap,   // [c][i][j]
                const __bf16* __restrict__ Bp,   // [c][j][k]
                const __bf16* __restrict__ Gp,   // [c][i][k]
                __bf16* __restrict__ U) {        // [(i*N+k)][c]
  __shared__ __align__(16) __bf16 As[2][128 * 40];
  __shared__ __align__(16) __bf16 Bs[2][32 * 72];
  const int tid  = threadIdx.x;
  const int lane = tid & 31, wave = tid >> 5;
  const int wm = wave >> 1, wn = wave & 1;
  const int i0 = blockIdx.x * 128;
  const int k0 = blockIdx.y * 64;
  const int c  = blockIdx.z;
  const __bf16* Ac = Ap + (size_t)c * NN;
  const __bf16* Bc = Bp + (size_t)c * NN;
  const __bf16* Gc = Gp + (size_t)c * NN;

  const int rA = tid >> 1, halfA = tid & 1;   // A tile: row, 16-col half
  const int rB = tid >> 3, segB = tid & 7;    // B tile: row, 8-col segment

  // 3 x b128 async copies per thread per j-step (8KB A tile + 4KB B tile)
  auto issue = [&](int buf, int j0) {
    const __bf16* ga = Ac + (size_t)(i0 + rA) * NDIM + j0 + halfA * 16;
    __bf16* la = &As[buf][rA * 40 + halfA * 16];
    async_copy16(ga,     la);
    async_copy16(ga + 8, la + 8);
    async_copy16(Bc + (size_t)(j0 + rB) * NDIM + k0 + segB * 8,
                 &Bs[buf][rB * 72 + segB * 8]);
  };

  v8f acc[2][2] = {};

  issue(0, 0);
  for (int j0 = 0; j0 < NDIM; j0 += 32) {
    const int cur = (j0 >> 5) & 1;
    if (j0 + 32 < NDIM) {           // prefetch next tile into other buffer
      issue(cur ^ 1, j0 + 32);
      wait_async<3>();              // retire current buffer's 3 copies only
    } else {
      wait_async<0>();
    }
    __syncthreads();

    v16bf a0 = load_a(As[cur], 40, wm * 32,      0, lane);
    v16bf a1 = load_a(As[cur], 40, wm * 32 + 16, 0, lane);
    v16bf b0 = load_b(Bs[cur], 72, 0, wn * 32,      lane);
    v16bf b1 = load_b(Bs[cur], 72, 0, wn * 32 + 16, lane);
    acc[0][0] = wmma_bf16(a0, b0, acc[0][0]);
    acc[0][1] = wmma_bf16(a0, b1, acc[0][1]);
    acc[1][0] = wmma_bf16(a1, b0, acc[1][0]);
    acc[1][1] = wmma_bf16(a1, b1, acc[1][1]);
    __syncthreads();
  }

  const int lm = lane & 15, hi = lane >> 4;
  #pragma unroll
  for (int mi = 0; mi < 2; ++mi)
    #pragma unroll
    for (int ni = 0; ni < 2; ++ni) {
      #pragma unroll
      for (int e = 0; e < 8; ++e) {
        int i = i0 + wm * 32 + mi * 16 + e + 8 * hi;
        int k = k0 + wn * 32 + ni * 16 + lm;
        float g = (float)Gc[(size_t)i * NDIM + k];
        float u = acc[mi][ni][e] * g;
        U[((size_t)i * NDIM + k) * DDIM + c] = (__bf16)u;
      }
    }
}

// ---------------------------------------------------------------------------
// Kernel 3: out = LayerNorm(pair_rep + U @ Wo + bo) * gamma + beta
// Block: 128 rows x full D=128 cols (needed for row-wise LN), 8 waves 4x2,
// each wave 32x64 (2x4 WMMA frags). Residual staged in LDS for LN.
// U tile (pure bf16 copy) goes through the async-to-LDS path.
// ---------------------------------------------------------------------------
__global__ __launch_bounds__(256)
void out_kernel(const __bf16* __restrict__ U,     // (NN, 128) bf16
                const float*  __restrict__ Wo,    // (128, 128)
                const float*  __restrict__ bo,    // (128)
                const float*  __restrict__ x,     // pair_rep (NN, 128)
                const float*  __restrict__ gamma,
                const float*  __restrict__ beta,
                float* __restrict__ out) {
  __shared__ __align__(16) __bf16 As[128 * 40];
  __shared__ __align__(16) __bf16 Ws[32 * 136];
  __shared__ float  res[128 * 132];
  const int tid  = threadIdx.x;
  const int lane = tid & 31, wave = tid >> 5;
  const int wm = wave >> 1, wn = wave & 1;   // wave tile: 32 rows x 64 cols
  const int m0 = blockIdx.x * 128;

  v8f acc[2][4] = {};

  for (int k0 = 0; k0 < DDIM; k0 += 32) {
    { // U tile 128x32 (already bf16): async copy to LDS
      int r = tid >> 1, half = tid & 1;
      const __bf16* src = U + (size_t)(m0 + r) * DDIM + k0 + half * 16;
      __bf16* dst = As + r * 40 + half * 16;
      async_copy16(src,     dst);
      async_copy16(src + 8, dst + 8);
    }
    { // Wo tile 32x128 f32 -> bf16 (conversion: VGPR path)
      int kr = tid >> 3, seg = tid & 7;
      const float* src = Wo + (size_t)(k0 + kr) * DDIM + seg * 16;
      Pack16 pk;
      #pragma unroll
      for (int i = 0; i < 16; i += 4) {
        v4f t = *(const v4f*)(src + i);
        pk.s[i] = (__bf16)t.x; pk.s[i+1] = (__bf16)t.y;
        pk.s[i+2] = (__bf16)t.z; pk.s[i+3] = (__bf16)t.w;
      }
      v8bf* dst = (v8bf*)(Ws + kr * 136 + seg * 16);
      dst[0] = pk.v[0]; dst[1] = pk.v[1];
    }
    wait_async<0>();
    __syncthreads();
    v16bf a0 = load_a(As, 40, wm * 32,      0, lane);
    v16bf a1 = load_a(As, 40, wm * 32 + 16, 0, lane);
    #pragma unroll
    for (int ni = 0; ni < 4; ++ni) {
      v16bf b = load_b(Ws, 136, 0, wn * 64 + ni * 16, lane);
      acc[0][ni] = wmma_bf16(a0, b, acc[0][ni]);
      acc[1][ni] = wmma_bf16(a1, b, acc[1][ni]);
    }
    __syncthreads();
  }

  const int lm = lane & 15, hi = lane >> 4;
  #pragma unroll
  for (int mi = 0; mi < 2; ++mi)
    #pragma unroll
    for (int ni = 0; ni < 4; ++ni) {
      int col = wn * 64 + ni * 16 + lm;
      float bv = bo[col];
      #pragma unroll
      for (int e = 0; e < 8; ++e) {
        int r = wm * 32 + mi * 16 + e + 8 * hi;
        float v = acc[mi][ni][e] + bv + x[(size_t)(m0 + r) * DDIM + col];
        res[r * 132 + col] = v;
      }
    }
  __syncthreads();

  if (tid < 128) {  // one thread per row: LayerNorm over D=128
    const float* row = res + tid * 132;
    float mu = 0.f;
    #pragma unroll 4
    for (int c = 0; c < DDIM; ++c) mu += row[c];
    mu *= (1.0f / DDIM);
    float var = 0.f;
    #pragma unroll 4
    for (int c = 0; c < DDIM; ++c) { float d = row[c] - mu; var += d * d; }
    var *= (1.0f / DDIM);
    float rs = rsqrtf(var + LN_EPS);
    float* o = out + (size_t)(m0 + tid) * DDIM;
    #pragma unroll 4
    for (int c = 0; c < DDIM; ++c)
      o[c] = (row[c] - mu) * rs * gamma[c] + beta[c];
  }
}

// ---------------------------------------------------------------------------
extern "C" void kernel_launch(void* const* d_in, const int* in_sizes, int n_in,
                              void* d_out, int out_size, void* d_ws, size_t ws_size,
                              hipStream_t stream) {
  const float* x     = (const float*)d_in[0];
  const float* Wa    = (const float*)d_in[1];
  const float* ba    = (const float*)d_in[2];
  const float* Wb    = (const float*)d_in[3];
  const float* bb    = (const float*)d_in[4];
  const float* Wg    = (const float*)d_in[5];
  const float* bg    = (const float*)d_in[6];
  const float* Wo    = (const float*)d_in[7];
  const float* bo    = (const float*)d_in[8];
  const float* gamma = (const float*)d_in[9];
  const float* beta  = (const float*)d_in[10];
  float* out = (float*)d_out;

  const size_t planeElems = (size_t)DDIM * NN;   // 33,554,432 bf16 = 64 MB each
  __bf16* Ap = (__bf16*)d_ws;
  __bf16* Bp = Ap + planeElems;
  __bf16* Gp = Bp + planeElems;
  __bf16* U  = Gp + planeElems;                  // total 256 MB of workspace

  dim3 blk(256);
  {
    dim3 grd(NN / 128, DDIM / 64, 1);            // (2048, 2)
    proj_kernel<<<grd, blk, 0, stream>>>(x, Wa, ba, Ap, 0);
    proj_kernel<<<grd, blk, 0, stream>>>(x, Wb, bb, Bp, 1);
    proj_kernel<<<grd, blk, 0, stream>>>(x, Wg, bg, Gp, 2);
  }
  {
    dim3 grd(NDIM / 128, NDIM / 64, DDIM);       // (4, 8, 128)
    tri_kernel<<<grd, blk, 0, stream>>>(Ap, Bp, Gp, U);
  }
  {
    dim3 grd(NN / 128, 1, 1);                    // (2048)
    out_kernel<<<grd, blk, 0, stream>>>(U, Wo, bo, x, gamma, beta, out);
  }
}
